// StandardAttention_61289183314183
// MI455X (gfx1250) — matrix operand order repaired and checked by reading, not compiled, not run on hardware
//
#include <hip/hip_runtime.h>
#include <hip/hip_bf16.h>

#define DIM      768
#define NH       12
#define HD       64
#define BATCH    16
#define SEQ      1024
#define M_TOTAL  (BATCH * SEQ)   /* 16384 */
#define NOUT_QKV (3 * DIM)       /* 2304  */
#define ATT_SCALE 0.125f         /* 64^-0.5 */

typedef __attribute__((ext_vector_type(16))) __bf16 v16bf;
typedef __attribute__((ext_vector_type(8)))  __bf16 v8bf;
typedef __attribute__((ext_vector_type(8)))  float  v8f;
typedef __attribute__((ext_vector_type(4)))  int    v4i;

union V16U { v16bf v; v8bf h[2]; };

#define AS1 __attribute__((address_space(1)))
#define AS3 __attribute__((address_space(3)))

#if defined(__gfx1250__) && __has_builtin(__builtin_amdgcn_global_load_async_to_lds_b128)
#define HAVE_ASYNC_LDS 1
#else
#define HAVE_ASYNC_LDS 0
#endif

__device__ __forceinline__ void wait_async0() {
#if defined(__gfx1250__)
#if __has_builtin(__builtin_amdgcn_s_wait_asynccnt)
    __builtin_amdgcn_s_wait_asynccnt(0);
#else
    asm volatile("s_wait_asynccnt 0x0" ::: "memory");
#endif
    asm volatile("" ::: "memory");
#endif
}
__device__ __forceinline__ void wait_async16() {
#if defined(__gfx1250__)
#if __has_builtin(__builtin_amdgcn_s_wait_asynccnt)
    __builtin_amdgcn_s_wait_asynccnt(16);
#else
    asm volatile("s_wait_asynccnt 0x10" ::: "memory");
#endif
    asm volatile("" ::: "memory");
#endif
}

// ---------------------------------------------------------------------------
// Conversion / packing kernels
// ---------------------------------------------------------------------------
__global__ void k_cvt_f32_bf16(const float* __restrict__ in, __bf16* __restrict__ out, int n) {
    int i = blockIdx.x * blockDim.x + threadIdx.x;
    int stride = gridDim.x * blockDim.x;
    for (; i < n; i += stride) out[i] = (__bf16)in[i];
}

// W [nout, nin] row-major -> B-fragment-native panels:
//   Wp[(kt*nout + n)*32 + kk] = W[n*nin + kt*32 + kk]   (kk = k within 32-panel)
// In the GEMM, a lane's B fragment (col n, K = kh*16..kh*16+15) is then one
// contiguous 32-byte read, and a 64-col k-panel is one contiguous 4 KB region.
__global__ void k_weight_panel(const float* __restrict__ W, __bf16* __restrict__ Wp,
                               int nout, int nin) {
    int t = blockIdx.x * blockDim.x + threadIdx.x;
    int total8 = (nout * nin) >> 3;
    int stride = gridDim.x * blockDim.x;
    for (; t < total8; t += stride) {
        int j = t << 3;
        int n = j / nin;
        int k = j - n * nin;
        int kt = k >> 5, kk = k & 31;
        const float* src = W + j;
        __bf16* dst = Wp + ((size_t)kt * nout + n) * 32 + kk;
#pragma unroll
        for (int e = 0; e < 8; ++e) dst[e] = (__bf16)src[e];
    }
}

// ---------------------------------------------------------------------------
// LDS-free GEMM core: wave computes a 32x32 C tile, 4 WMMA per 32-deep k-step.
// A: row-major bf16 [M, DIM]; Wp: panel layout (see above).  Everything is
// L2-resident (192 MB L2 >> working set), so no LDS staging / barriers.
// ---------------------------------------------------------------------------
__device__ __forceinline__ void gemm_32x32(const __bf16* __restrict__ A,
                                           const __bf16* __restrict__ Wp, int nout,
                                           int mbase, int nbase, int l15, int kh,
                                           v8f acc[2][2]) {
    const int KT = DIM / 32;  // 24
#pragma unroll 2
    for (int kt = 0; kt < KT; ++kt) {
        V16U a0, a1, b0, b1;
        const __bf16* ap0 = A + (size_t)(mbase + l15) * DIM + kt * 32;
        a0.h[0] = *(const v8bf*)(ap0 + kh * 16);
        a0.h[1] = *(const v8bf*)(ap0 + kh * 16 + 8);
        const __bf16* ap1 = ap0 + (size_t)16 * DIM;
        a1.h[0] = *(const v8bf*)(ap1 + kh * 16);
        a1.h[1] = *(const v8bf*)(ap1 + kh * 16 + 8);
        const __bf16* bp = Wp + ((size_t)kt * nout + nbase + l15) * 32 + kh * 16;
        b0.v = *(const v16bf*)bp;
        b1.v = *(const v16bf*)(bp + 16 * 32);
        acc[0][0] = __builtin_amdgcn_wmma_f32_16x16x32_bf16(false, a0.v, false, b0.v,
                                                            (short)0, acc[0][0], false, false);
        acc[0][1] = __builtin_amdgcn_wmma_f32_16x16x32_bf16(false, a0.v, false, b1.v,
                                                            (short)0, acc[0][1], false, false);
        acc[1][0] = __builtin_amdgcn_wmma_f32_16x16x32_bf16(false, a1.v, false, b0.v,
                                                            (short)0, acc[1][0], false, false);
        acc[1][1] = __builtin_amdgcn_wmma_f32_16x16x32_bf16(false, a1.v, false, b1.v,
                                                            (short)0, acc[1][1], false, false);
    }
}

// ---------------------------------------------------------------------------
// QKV GEMM: C[16384, 2304] = Xb @ Wqkv^T ; scatter into Q, K (natural [b,h,n,d])
// and V transposed ([b,h,d,n]) bf16 buffers.
// Block = 256 threads = 8 waves arranged 4(M) x 2(N): block tile 128 x 64.
// ---------------------------------------------------------------------------
__global__ __launch_bounds__(256)
void k_qkv_gemm(const __bf16* __restrict__ X, const __bf16* __restrict__ Wp,
                __bf16* __restrict__ Q, __bf16* __restrict__ K,
                __bf16* __restrict__ Vt) {
    const int nbn = NOUT_QKV / 64;   // 36
    int bm = blockIdx.x / nbn, bn = blockIdx.x - bm * nbn;
    int wave = threadIdx.x >> 5, lane = threadIdx.x & 31;
    int wr = wave >> 1, wc = wave & 1;
    int l15 = lane & 15, kh = lane >> 4;
    int mbase = bm * 128 + wr * 32;
    int nbase = bn * 64 + wc * 32;

    v8f acc[2][2] = {};
    gemm_32x32(X, Wp, NOUT_QKV, mbase, nbase, l15, kh, acc);

#pragma unroll
    for (int tn = 0; tn < 2; ++tn) {
        int col = nbase + tn * 16 + l15;      // uniform (which,h) across the wave tile
        int which = col / DIM;
        int rem = col - which * DIM;
        int h = rem >> 6, d = rem & 63;
#pragma unroll
        for (int tm = 0; tm < 2; ++tm) {
#pragma unroll
            for (int r = 0; r < 8; ++r) {
                int m = mbase + tm * 16 + kh * 8 + r;
                int b = m >> 10, n = m & 1023;
                int hh = b * NH + h;
                __bf16 val = (__bf16)acc[tm][tn][r];
                if (which == 0)      Q[((size_t)hh * SEQ + n) * HD + d] = val;
                else if (which == 1) K[((size_t)hh * SEQ + n) * HD + d] = val;
                else                 Vt[((size_t)hh * HD + d) * SEQ + n] = val;
            }
        }
    }
}

// ---------------------------------------------------------------------------
// Flash attention: one wave per (head, 16-query block).  Online softmax.
// With async-to-LDS support: K/V 4 KB tiles double-buffered in LDS, next
// key-block prefetched with global_load_async_to_lds_b128 while the current
// one is computed; ASYNCcnt gates visibility (in-order completion).
// ---------------------------------------------------------------------------
__global__ __launch_bounds__(32)
void k_attention(const __bf16* __restrict__ Q, const __bf16* __restrict__ K,
                 const __bf16* __restrict__ Vt, __bf16* __restrict__ AttnOut) {
    __shared__ __align__(32) __bf16 sP[16 * 32];
#if HAVE_ASYNC_LDS
    __shared__ __align__(32) __bf16 sK[2][32 * 64];   // [key][d]
    __shared__ __align__(32) __bf16 sV[2][64 * 32];   // [d][key]
#endif
    int hh = blockIdx.x >> 6;                 // head index 0..191
    int m0 = (blockIdx.x & 63) * 16;          // query block
    int b = hh / NH, h = hh - b * NH;
    int lane = threadIdx.x & 31;
    int l15 = lane & 15, kh = lane >> 4;
    const __bf16* Qh = Q + (size_t)hh * SEQ * HD;
    const __bf16* Kh = K + (size_t)hh * SEQ * HD;
    const __bf16* Vh = Vt + (size_t)hh * HD * SEQ;

    // Q A-fragments for d-slices [0,32) and [32,64): contiguous per lane.
    V16U aq0, aq1;
    const __bf16* qrow = Qh + (size_t)(m0 + l15) * HD;
    aq0.h[0] = *(const v8bf*)(qrow + kh * 16);
    aq0.h[1] = *(const v8bf*)(qrow + kh * 16 + 8);
    aq1.h[0] = *(const v8bf*)(qrow + 32 + kh * 16);
    aq1.h[1] = *(const v8bf*)(qrow + 40 + kh * 16);

    v8f o0 = {}, o1 = {}, o2 = {}, o3 = {};
    float mrow[8], lrow[8];
#pragma unroll
    for (int r = 0; r < 8; ++r) { mrow[r] = -__builtin_inff(); lrow[r] = 0.f; }

#if HAVE_ASYNC_LDS
    auto issue_async = [&](int kb, int buf) {
        // K block: rows kb..kb+31 x 64 d = one contiguous 4 KB region.
        const __bf16* gk = Kh + (size_t)kb * HD;
#pragma unroll
        for (int i = 0; i < 8; ++i) {
            int q = i * 32 + lane;            // 0..255 b128 chunks
            __builtin_amdgcn_global_load_async_to_lds_b128(
                (AS1 v4i*)(gk + q * 8),
                (AS3 v4i*)(&sK[buf][q * 8]), 0, 0);
        }
        // V block: 64 d-rows x 32 keys, rows strided by SEQ in Vt.
#pragma unroll
        for (int i = 0; i < 8; ++i) {
            int q = i * 32 + lane;
            int row = q >> 2, c = (q & 3) * 8;
            __builtin_amdgcn_global_load_async_to_lds_b128(
                (AS1 v4i*)(Vh + (size_t)row * SEQ + kb + c),
                (AS3 v4i*)(&sV[buf][row * 32 + c]), 0, 0);
        }
    };
    issue_async(0, 0);
    int cur = 0;
#endif

    for (int kb = 0; kb < SEQ; kb += 32) {
#if HAVE_ASYNC_LDS
        bool havenext = (kb + 32 < SEQ);
        if (havenext) issue_async(kb + 32, cur ^ 1);
        if (havenext) wait_async16(); else wait_async0();
        const __bf16* kp0 = &sK[cur][(size_t)l15 * HD];
        const __bf16* kp1 = &sK[cur][(size_t)(16 + l15) * HD];
#else
        const __bf16* kp0 = Kh + (size_t)(kb + l15) * HD;
        const __bf16* kp1 = Kh + (size_t)(kb + 16 + l15) * HD;
#endif
        // ---- scores: two 16x16 key tiles -------------------------------
        v8f s0 = {}, s1 = {};
        {
            V16U bk;
            bk.v = *(const v16bf*)(kp0 + kh * 16);
            s0 = __builtin_amdgcn_wmma_f32_16x16x32_bf16(false, aq0.v, false, bk.v,
                                                         (short)0, s0, false, false);
            bk.v = *(const v16bf*)(kp0 + 32 + kh * 16);
            s0 = __builtin_amdgcn_wmma_f32_16x16x32_bf16(false, aq1.v, false, bk.v,
                                                         (short)0, s0, false, false);
            bk.v = *(const v16bf*)(kp1 + kh * 16);
            s1 = __builtin_amdgcn_wmma_f32_16x16x32_bf16(false, aq0.v, false, bk.v,
                                                         (short)0, s1, false, false);
            bk.v = *(const v16bf*)(kp1 + 32 + kh * 16);
            s1 = __builtin_amdgcn_wmma_f32_16x16x32_bf16(false, aq1.v, false, bk.v,
                                                         (short)0, s1, false, false);
        }
        // ---- online softmax (C layout: lane holds rows kh*8+r, col l15) ----
#pragma unroll
        for (int r = 0; r < 8; ++r) {
            float x0 = s0[r] * ATT_SCALE, x1 = s1[r] * ATT_SCALE;
            float mb = fmaxf(x0, x1);
#pragma unroll
            for (int off = 8; off >= 1; off >>= 1)
                mb = fmaxf(mb, __shfl_xor(mb, off, 32));
            float mnew = fmaxf(mrow[r], mb);
            float corr = __expf(mrow[r] - mnew);
            float p0 = __expf(x0 - mnew);
            float p1 = __expf(x1 - mnew);
            float ps = p0 + p1;
#pragma unroll
            for (int off = 8; off >= 1; off >>= 1)
                ps += __shfl_xor(ps, off, 32);
            lrow[r] = lrow[r] * corr + ps;
            mrow[r] = mnew;
            o0[r] *= corr; o1[r] *= corr; o2[r] *= corr; o3[r] *= corr;
            int row = kh * 8 + r;
            sP[row * 32 + l15]      = (__bf16)p0;
            sP[row * 32 + 16 + l15] = (__bf16)p1;
        }
        __syncthreads();
        // ---- P: C layout -> A layout via LDS ---------------------------
        V16U pf;
        const __bf16* pp = sP + l15 * 32;
        pf.h[0] = *(const v8bf*)(pp + kh * 16);
        pf.h[1] = *(const v8bf*)(pp + kh * 16 + 8);
        // ---- O += P V ---------------------------------------------------
#pragma unroll
        for (int dt = 0; dt < 4; ++dt) {
            V16U bv;
#if HAVE_ASYNC_LDS
            bv.v = *(const v16bf*)(&sV[cur][(dt * 16 + l15) * 32 + kh * 16]);
#else
            bv.v = *(const v16bf*)(Vh + (size_t)(dt * 16 + l15) * SEQ + kb + kh * 16);
#endif
            v8f& o = dt == 0 ? o0 : dt == 1 ? o1 : dt == 2 ? o2 : o3;
            o = __builtin_amdgcn_wmma_f32_16x16x32_bf16(false, pf.v, false, bv.v,
                                                        (short)0, o, false, false);
        }
        __syncthreads();
#if HAVE_ASYNC_LDS
        cur ^= 1;
#endif
    }

    // ---- normalize + store [B, N, C] bf16 for the projection GEMM ------
    float rcp[8];
#pragma unroll
    for (int r = 0; r < 8; ++r) rcp[r] = 1.0f / lrow[r];
#pragma unroll
    for (int r = 0; r < 8; ++r) {
        int n = m0 + kh * 8 + r;
        size_t base = ((size_t)b * SEQ + n) * DIM + h * HD;
        AttnOut[base +      l15] = (__bf16)(o0[r] * rcp[r]);
        AttnOut[base + 16 + l15] = (__bf16)(o1[r] * rcp[r]);
        AttnOut[base + 32 + l15] = (__bf16)(o2[r] * rcp[r]);
        AttnOut[base + 48 + l15] = (__bf16)(o3[r] * rcp[r]);
    }
}

// ---------------------------------------------------------------------------
// Output projection: out[16384, 768] = AttnOut @ Wp^T + bias  (f32 out)
// ---------------------------------------------------------------------------
__global__ __launch_bounds__(256)
void k_proj_gemm(const __bf16* __restrict__ A, const __bf16* __restrict__ Wp,
                 const float* __restrict__ bias, float* __restrict__ out) {
    const int nbn = DIM / 64;        // 12
    int bm = blockIdx.x / nbn, bn = blockIdx.x - bm * nbn;
    int wave = threadIdx.x >> 5, lane = threadIdx.x & 31;
    int wr = wave >> 1, wc = wave & 1;
    int l15 = lane & 15, kh = lane >> 4;
    int mbase = bm * 128 + wr * 32;
    int nbase = bn * 64 + wc * 32;

    v8f acc[2][2] = {};
    gemm_32x32(A, Wp, DIM, mbase, nbase, l15, kh, acc);

#pragma unroll
    for (int tn = 0; tn < 2; ++tn) {
        int col = nbase + tn * 16 + l15;
        float bval = bias[col];
#pragma unroll
        for (int tm = 0; tm < 2; ++tm) {
#pragma unroll
            for (int r = 0; r < 8; ++r) {
                int m = mbase + tm * 16 + kh * 8 + r;
                out[(size_t)m * DIM + col] = acc[tm][tn][r] + bval;
            }
        }
    }
}

// ---------------------------------------------------------------------------
extern "C" void kernel_launch(void* const* d_in, const int* in_sizes, int n_in,
                              void* d_out, int out_size, void* d_ws, size_t ws_size,
                              hipStream_t stream) {
    const float* x      = (const float*)d_in[0];
    const float* w_qkv  = (const float*)d_in[1];
    const float* w_proj = (const float*)d_in[2];
    const float* b_proj = (const float*)d_in[3];
    float* out = (float*)d_out;

    char* ws = (char*)d_ws;
    size_t off = 0;
    auto alloc = [&](size_t bytes) {
        char* p = ws + off;
        off += (bytes + 255) & ~(size_t)255;
        return p;
    };
    const size_t EL = (size_t)M_TOTAL * DIM;            // 12.58M elements
    __bf16* Xb  = (__bf16*)alloc(EL * 2);
    __bf16* Wqp = (__bf16*)alloc((size_t)DIM * NOUT_QKV * 2);
    __bf16* Wpp = (__bf16*)alloc((size_t)DIM * DIM * 2);
    __bf16* Qb  = (__bf16*)alloc(EL * 2);
    __bf16* Kb  = (__bf16*)alloc(EL * 2);
    __bf16* Vtb = (__bf16*)alloc(EL * 2);
    __bf16* AOb = (__bf16*)alloc(EL * 2);

    k_cvt_f32_bf16<<<4096, 256, 0, stream>>>(x, Xb, (int)EL);
    k_weight_panel<<<1024, 256, 0, stream>>>(w_qkv, Wqp, NOUT_QKV, DIM);
    k_weight_panel<<<512, 256, 0, stream>>>(w_proj, Wpp, DIM, DIM);

    k_qkv_gemm<<<(M_TOTAL / 128) * (NOUT_QKV / 64), 256, 0, stream>>>(Xb, Wqp, Qb, Kb, Vtb);

    k_attention<<<BATCH * NH * (SEQ / 16), 32, 0, stream>>>(Qb, Kb, Vtb, AOb);

    k_proj_gemm<<<(M_TOTAL / 128) * (DIM / 64), 256, 0, stream>>>(AOb, Wpp, b_proj, out);
}
